// IMUDecoderBlock_76244259439359
// MI455X (gfx1250) — compile-verified, matrix operand order checked by reading, not run on hardware
//
#include <hip/hip_runtime.h>

// ---------------------------------------------------------------------------
// IMU decoder block for MI455X (gfx1250, wave32, WMMA 16x16x32 f16->f32).
// All GEMMs and both attention matmuls run on v_wmma_f32_16x16x32_f16.
// GEMM: template-specialized epilogue, double-buffered LDS software pipeline,
// global_prefetch_b8 ahead. Attention: flash-style online softmax whose
// 16-lane row reductions use DPP16 (VALU) instead of ds_bpermute.
// ---------------------------------------------------------------------------

typedef __attribute__((ext_vector_type(16))) _Float16 v16h;
typedef __attribute__((ext_vector_type(8)))  _Float16 v8h;
typedef __attribute__((ext_vector_type(4)))  _Float16 v4h;
typedef __attribute__((ext_vector_type(8)))  float    v8f;

#define D_MODEL  768
#define N_HEADS  12
#define HEAD_DIM 64
#define GLD      40   // padded f16 row stride in LDS tiles (80B -> bank spread)

union HFrag { v16h v; v8h h8[2]; };

__device__ __forceinline__ v8f wmma_16x16x32(v16h a, v16h b, v8f c) {
  return __builtin_amdgcn_wmma_f32_16x16x32_f16(false, a, false, b, (short)0, c,
                                                false, false);
}

// DPP16 cross-lane fetch (VALU, no LDS). CTRL: 0xB1 quad_perm xor1,
// 0x4E quad_perm xor2, 0x141 row_half_mirror, 0x140 row_mirror.
template <int CTRL>
__device__ __forceinline__ float dppmov(float x) {
  return __int_as_float(__builtin_amdgcn_update_dpp(
      0, __float_as_int(x), CTRL, 0xF, 0xF, true));
}
// Reduce across the 16-lane DPP row (our C-layout row group).
__device__ __forceinline__ float redmax16(float v) {
  v = fmaxf(v, dppmov<0xB1>(v));
  v = fmaxf(v, dppmov<0x4E>(v));
  v = fmaxf(v, dppmov<0x141>(v));
  v = fmaxf(v, dppmov<0x140>(v));
  return v;
}
__device__ __forceinline__ float redsum16(float v) {
  v += dppmov<0xB1>(v);
  v += dppmov<0x4E>(v);
  v += dppmov<0x141>(v);
  v += dppmov<0x140>(v);
  return v;
}

// Load a 16x32 f16 operand fragment in the CDNA5 A layout:
//   lanes 0-15 : row = row0+lane,    halves [0..7] and [16..23]
//   lanes16-31 : row = row0+lane-16, halves [8..15] and [24..31]
// The same layout serves our B operand: lane n holds column n of B (= row n of
// W / K / V^T, K-major), since W is stored [N,K] row-major.
__device__ __forceinline__ v16h ldfrag(const _Float16* __restrict__ base,
                                       int row0, int ld) {
  const int lane = threadIdx.x & 31;
  const _Float16* p = base + (size_t)(row0 + (lane & 15)) * ld + ((lane >> 4) << 3);
  HFrag u;
  u.h8[0] = *(const v8h*)p;
  u.h8[1] = *(const v8h*)(p + 16);
  return u.v;
}

// ---------------------------------------------------------------------------
// LayerNorm over 768 features, one block per token.
// ---------------------------------------------------------------------------
__global__ __launch_bounds__(256) void ln768_kernel(
    const float* __restrict__ x, const float* __restrict__ g,
    const float* __restrict__ b, float* __restrict__ o) {
  __shared__ float red[256];
  const int t = blockIdx.x, tid = threadIdx.x;
  const float* xp = x + (size_t)t * D_MODEL;
  float v0 = xp[tid], v1 = xp[tid + 256], v2 = xp[tid + 512];
  red[tid] = v0 + v1 + v2;
  __syncthreads();
  for (int off = 128; off; off >>= 1) {
    if (tid < off) red[tid] += red[tid + off];
    __syncthreads();
  }
  const float mean = red[0] * (1.f / 768.f);
  __syncthreads();
  const float d0 = v0 - mean, d1 = v1 - mean, d2 = v2 - mean;
  red[tid] = d0 * d0 + d1 * d1 + d2 * d2;
  __syncthreads();
  for (int off = 128; off; off >>= 1) {
    if (tid < off) red[tid] += red[tid + off];
    __syncthreads();
  }
  const float rstd = rsqrtf(red[0] * (1.f / 768.f) + 1e-5f);
  float* op = o + (size_t)t * D_MODEL;
  op[tid]       = d0 * rstd * g[tid]       + b[tid];
  op[tid + 256] = d1 * rstd * g[tid + 256] + b[tid + 256];
  op[tid + 512] = d2 * rstd * g[tid + 512] + b[tid + 512];
}

// ---------------------------------------------------------------------------
// Generic GEMM: C[M,N] = act(A[M,K] * W[N,K]^T [+bias]) [+res].
// 128x128 workgroup tile, K staged in 32-chunks as f16 in double-buffered LDS.
// 8 waves, each computes a 64x32 tile = 8 WMMA accumulators.
// Pipeline: global->reg fetch of tile k+1 issued before the WMMAs of tile k;
// ds_store into the ping-pong buffer afterwards; one barrier per iteration.
// M%128==0, N%128==0, K%32==0 guaranteed by the caller.
// ---------------------------------------------------------------------------
template <bool BIAS, bool RES, bool GELU>
__global__ __launch_bounds__(256) void gemm_wmma_kernel(
    const float* __restrict__ A, int lda,
    const float* __restrict__ W,               // [N,K] row-major
    const float* __restrict__ bias,            // [N]    (BIAS)
    const float* __restrict__ res, int ldres,  // [M,*]  (RES)
    float* __restrict__ C, int ldc,
    int K) {
  __shared__ _Float16 Asl[2][128 * GLD];
  __shared__ _Float16 Wsl[2][128 * GLD];
  const int tid = threadIdx.x;
  const int lane = tid & 31;
  const int wave = tid >> 5;
  const int wm = wave & 1;   // 0..1 -> 64 rows each
  const int wn = wave >> 1;  // 0..3 -> 32 cols each
  const size_t mBase = (size_t)blockIdx.y * 128;
  const size_t nBase = (size_t)blockIdx.x * 128;

  // Per-thread staging slots: 4 float4 from A, 4 float4 from W per K-tile.
  const int srow = tid >> 3;              // 0..31 (row stride 32 over 4 iters)
  const int sc4 = (tid & 7) << 2;         // 0..28
  const float* ga = A + (mBase + srow) * (size_t)lda + sc4;
  const float* gw = W + (nBase + srow) * (size_t)K + sc4;

  float4 avr[4], wvr[4];
  auto gload = [&](int k0) {
#pragma unroll
    for (int it = 0; it < 4; ++it) {
      avr[it] = *(const float4*)(ga + (size_t)(it * 32) * lda + k0);
      wvr[it] = *(const float4*)(gw + (size_t)(it * 32) * K + k0);
    }
    // Warm L2/WGP$ one K-tile further ahead (global_prefetch_b8).
    if (k0 + 32 < K) {
      __builtin_prefetch(ga + k0 + 32, 0, 1);
      __builtin_prefetch(gw + k0 + 32, 0, 1);
    }
  };
  auto sstore = [&](int buf) {
#pragma unroll
    for (int it = 0; it < 4; ++it) {
      v4h ah = {(_Float16)avr[it].x, (_Float16)avr[it].y, (_Float16)avr[it].z,
                (_Float16)avr[it].w};
      v4h wh = {(_Float16)wvr[it].x, (_Float16)wvr[it].y, (_Float16)wvr[it].z,
                (_Float16)wvr[it].w};
      *(v4h*)(Asl[buf] + (srow + it * 32) * GLD + sc4) = ah;
      *(v4h*)(Wsl[buf] + (srow + it * 32) * GLD + sc4) = wh;
    }
  };

  v8f acc[4][2] = {};
  gload(0);
  sstore(0);
  const int nK = K >> 5;
  for (int kt = 0; kt < nK; ++kt) {
    __syncthreads();                       // tile kt visible to all waves
    const int cur = kt & 1;
    const bool more = (kt + 1) < nK;
    if (more) gload((kt + 1) << 5);        // overlap with WMMAs below
    const v16h bf0 = ldfrag(Wsl[cur], wn * 32, GLD);
    const v16h bf1 = ldfrag(Wsl[cur], wn * 32 + 16, GLD);
#pragma unroll
    for (int i = 0; i < 4; ++i) {
      const v16h af = ldfrag(Asl[cur], wm * 64 + i * 16, GLD);
      acc[i][0] = wmma_16x16x32(af, bf0, acc[i][0]);
      acc[i][1] = wmma_16x16x32(af, bf1, acc[i][1]);
    }
    if (more) sstore(cur ^ 1);             // other buffer: no race with reads
  }

  // Epilogue. C layout: VGPR r, lanes0-15 -> (M=r, N=lane); lanes16-31 -> (M=r+8).
  const int rsel = (lane >> 4) << 3;
  const int ncol = lane & 15;
#pragma unroll
  for (int i = 0; i < 4; ++i)
#pragma unroll
    for (int j = 0; j < 2; ++j) {
      const size_t n = nBase + wn * 32 + j * 16 + ncol;
      const float bv = BIAS ? bias[n] : 0.f;
#pragma unroll
      for (int r = 0; r < 8; ++r) {
        const size_t m = mBase + wm * 64 + i * 16 + rsel + r;
        float v = acc[i][j][r] + bv;
        if (GELU) v = 0.5f * v * (1.f + erff(v * 0.70710678118654752440f));
        if (RES) v += res[m * (size_t)ldres + n];
        C[m * (size_t)ldc + n] = v;
      }
    }
}

// ---------------------------------------------------------------------------
// RoPE2D (croco): head_dim split into (y,x) 32-halves; half-split rotation
// with 16 freqs per half, base=100. One thread per rotation pair (384/token).
// ---------------------------------------------------------------------------
__global__ __launch_bounds__(384) void rope2d_pack_kernel(
    const float* __restrict__ src, int ldsrc, const int* __restrict__ pos,
    _Float16* __restrict__ dst, int S) {
  const int t = blockIdx.x;                  // token = b*S + s
  const int tid = threadIdx.x;               // 0..383
  const int head = tid >> 5;
  const int rem = tid & 31;
  const int half = rem >> 4;
  const int j = rem & 15;
  const int b = t / S, s = t - b * S;
  const float p = (float)pos[t * 2 + half];
  const float fr = p * __powf(100.f, -(float)j / 16.f);
  float sn, cs;
  __sincosf(fr, &sn, &cs);
  const float* sp = src + (size_t)t * ldsrc + head * HEAD_DIM + half * 32;
  const float t1 = sp[j], t2 = sp[j + 16];
  _Float16* dp = dst + ((size_t)(b * N_HEADS + head) * S + s) * HEAD_DIM + half * 32;
  dp[j]      = (_Float16)(t1 * cs - t2 * sn);
  dp[j + 16] = (_Float16)(t2 * cs + t1 * sn);
}

// RoPE1D (lucidrains): interleaved pairs, pos = arange(S), base=10000.
__global__ __launch_bounds__(384) void rope1d_pack_kernel(
    const float* __restrict__ src, int ldsrc, _Float16* __restrict__ dst, int S) {
  const int t = blockIdx.x;
  const int tid = threadIdx.x;
  const int head = tid >> 5;
  const int i = tid & 31;
  const int b = t / S, s = t - b * S;
  const float fr = (float)s * __powf(10000.f, -(float)i / 32.f);
  float sn, cs;
  __sincosf(fr, &sn, &cs);
  const float* sp = src + (size_t)t * ldsrc + head * HEAD_DIM;
  const float t1 = sp[2 * i], t2 = sp[2 * i + 1];
  _Float16* dp = dst + ((size_t)(b * N_HEADS + head) * S + s) * HEAD_DIM;
  dp[2 * i]     = (_Float16)(t1 * cs - t2 * sn);
  dp[2 * i + 1] = (_Float16)(t2 * cs + t1 * sn);
}

// Pack V transposed: dst[B,H,64,S] f16 <- src[T,*] f32 (unpermuted heads).
__global__ __launch_bounds__(768) void pack_vt_kernel(
    const float* __restrict__ src, int ldsrc, _Float16* __restrict__ dst, int S) {
  const int t = blockIdx.x;
  const int tid = threadIdx.x;
  const int head = tid >> 6;
  const int dd = tid & 63;
  const int b = t / S, s = t - b * S;
  dst[((size_t)(b * N_HEADS + head) * HEAD_DIM + dd) * S + s] =
      (_Float16)src[(size_t)t * ldsrc + head * HEAD_DIM + dd];
}

// ---------------------------------------------------------------------------
// Flash attention, one 16-row Q tile per wave, online softmax.
// Q,K: [B*H, S, 64] f16 (roped). Vt: [B*H, 64, Sk] f16.
// Out: f32 "unheads" buffer [B*Sq, outLd], written at col outOff + h*64.
// Sq % 128 == 0, Sk % 32 == 0.
// Row stats live per 16-lane DPP row; reductions are pure VALU (DPP16),
// which co-executes with the XDL WMMA pipe.
// ---------------------------------------------------------------------------
__global__ __launch_bounds__(256) void flash_attn_kernel(
    const _Float16* __restrict__ Q, const _Float16* __restrict__ Kt,
    const _Float16* __restrict__ Vt, float* __restrict__ Out, int outLd,
    int outOff, int Sq, int Sk) {
  __shared__ _Float16 Pl[8][16 * GLD];   // per-wave P relayout strip
  const int lane = threadIdx.x & 31;
  const int wave = threadIdx.x >> 5;
  const int bh = blockIdx.y;
  const int b = bh / N_HEADS;
  const int h = bh - b * N_HEADS;
  const int m0 = (blockIdx.x * 8 + wave) * 16;
  const _Float16* Qb = Q + (size_t)bh * Sq * HEAD_DIM;
  const _Float16* Kb = Kt + (size_t)bh * Sk * HEAD_DIM;
  const _Float16* Vb = Vt + (size_t)bh * HEAD_DIM * Sk;

  const v16h qf0 = ldfrag(Qb, m0, HEAD_DIM);        // dims 0..31
  const v16h qf1 = ldfrag(Qb + 32, m0, HEAD_DIM);   // dims 32..63

  v8f accO[4] = {};
  float mrow[8], lrow[8];
#pragma unroll
  for (int r = 0; r < 8; ++r) { mrow[r] = -3.0e38f; lrow[r] = 0.f; }
  const int rsel = (lane >> 4) << 3;   // this half-wave owns rows rsel..rsel+7
  _Float16* pl = Pl[wave];

  for (int kc = 0; kc < Sk; kc += 32) {
    if (kc + 32 < Sk) {  // warm next key/value chunk (global_prefetch_b8)
      __builtin_prefetch(Kb + (size_t)(kc + 32 + (lane & 15)) * HEAD_DIM, 0, 1);
      __builtin_prefetch(Vb + (size_t)(lane & 15) * Sk + kc + 32, 0, 1);
    }
    v8f s0 = {}, s1 = {};
    {
      v16h kf0 = ldfrag(Kb, kc, HEAD_DIM);
      v16h kf1 = ldfrag(Kb + 32, kc, HEAD_DIM);
      s0 = wmma_16x16x32(qf0, kf0, s0);
      s0 = wmma_16x16x32(qf1, kf1, s0);
      kf0 = ldfrag(Kb, kc + 16, HEAD_DIM);
      kf1 = ldfrag(Kb + 32, kc + 16, HEAD_DIM);
      s1 = wmma_16x16x32(qf0, kf0, s1);
      s1 = wmma_16x16x32(qf1, kf1, s1);
    }
    s0 = s0 * 0.125f;   // 1/sqrt(64)
    s1 = s1 * 0.125f;
#pragma unroll
    for (int r = 0; r < 8; ++r) {
      const float nm = fmaxf(mrow[r], redmax16(fmaxf(s0[r], s1[r])));
      const float p0 = __expf(s0[r] - nm);
      const float p1 = __expf(s1[r] - nm);
      const float rs = redsum16(p0 + p1);
      const float corr = __expf(mrow[r] - nm);
      lrow[r] = lrow[r] * corr + rs;
      mrow[r] = nm;
#pragma unroll
      for (int j = 0; j < 4; ++j) accO[j][r] = accO[j][r] * corr;
      // C layout -> A layout via per-wave LDS (in-order LDS per wave)
      pl[(rsel + r) * GLD + (lane & 15)]      = (_Float16)p0;
      pl[(rsel + r) * GLD + 16 + (lane & 15)] = (_Float16)p1;
    }
    const v16h pf = ldfrag(pl, 0, GLD);
#pragma unroll
    for (int j = 0; j < 4; ++j) {
      const v16h vf = ldfrag(Vb + kc, j * 16, Sk);
      accO[j] = wmma_16x16x32(pf, vf, accO[j]);
    }
  }

  float* ob = Out + (size_t)(b * Sq + m0 + rsel) * outLd + outOff + h * HEAD_DIM;
#pragma unroll
  for (int r = 0; r < 8; ++r) {
    const float invl = 1.f / lrow[r];
#pragma unroll
    for (int j = 0; j < 4; ++j)
      ob[(size_t)r * outLd + j * 16 + (lane & 15)] = accO[j][r] * invl;
  }
}

// ---------------------------------------------------------------------------
// Host-side orchestration.
// ---------------------------------------------------------------------------
extern "C" void kernel_launch(void* const* d_in, const int* in_sizes, int n_in,
                              void* d_out, int out_size, void* d_ws,
                              size_t ws_size, hipStream_t stream) {
  (void)in_sizes; (void)n_in; (void)out_size; (void)ws_size;
  const int B = 4, Sm = 1024, Si = 512, Su = 512;
  const int Tm = B * Sm, Ti = B * Si, Tu = B * Su;
  const int D = D_MODEL, HID = 3072;

  const float* masked   = (const float*)d_in[0];
  const int*   mpos     = (const int*)d_in[1];
  const float* imu      = (const float*)d_in[2];
  const float* unmasked = (const float*)d_in[3];
  const int*   upos     = (const int*)d_in[4];
#define PF(i) ((const float*)d_in[(i)])
  // param indices (dict insertion order):
  // 5 norm_y_g 6 norm_y_b 7 norm1_g 8 norm1_b 9 norm2_g 10 norm2_b
  // 11 norm3_g 12 norm3_b 13 norm4_g 14 norm4_b 15 norm5_g 16 norm5_b
  // 17 norm6_g 18 norm6_b 19 sa1_qkv 20 sa1_pw 21 sa1_pb 22 sa2_qkv
  // 23 sa2_pw 24 sa2_pb 25 ca1_q 26 ca1_k 27 ca1_v 28 ca1_pw 29 ca1_pb
  // 30 ca2_q 31 ca2_v 32 ca2_pw 33 ca2_pb 34 ca3_q 35 ca3_v 36 ca3_pw
  // 37 ca3_pb 38 ca4_q 39 ca4_v 40 ca4_pw 41 ca4_pb 42 fuse1_w 43 fuse1_b
  // 44 fuse2_w 45 fuse2_b 46 mlp1_w1 47 mlp1_b1 48 mlp1_w2 49 mlp1_b2
  // 50 mlp2_w1 51 mlp2_b1 52 mlp2_w2 53 mlp2_b2

  float* out_x  = (float*)d_out;                 // x   [4,1024,768]
  float* out_y  = out_x + (size_t)Tm * D;        // y   [4,512,768]
  float* out_m2 = out_y + (size_t)Ti * D;        // masked after self_attn1

  char* ws = (char*)d_ws;
  size_t off = 0;
  auto alloc = [&](size_t bytes) -> char* {
    char* p = ws + off;
    off += (bytes + 255) & ~(size_t)255;
    return p;
  };
  float* f_unm  = (float*)alloc((size_t)Tu * D * 4);       // LN(unmasked)
  float* f_mn   = (float*)alloc((size_t)Tm * D * 4);       // norm3(masked2)
  float* f_inn  = (float*)alloc((size_t)Ti * D * 4);       // norm4(imu2)
  float* f_ln   = (float*)alloc((size_t)Tm * D * 4);       // transient LN out
  float* f_big  = (float*)alloc((size_t)Tm * HID * 4);     // qkv / mlp hidden
  float* f_p1   = (float*)alloc((size_t)Tm * D * 4);       // projection scratch
  float* f_p2   = (float*)alloc((size_t)Tm * D * 4);       // projection scratch
  float* f_ao   = (float*)alloc((size_t)Tm * D * 4);       // attn output
  float* f_imu2 = (float*)alloc((size_t)Ti * D * 4);       // imu after sa2
  float* f_x12  = (float*)alloc((size_t)Tm * 2 * D * 4);   // concat(x1,x2)
  float* f_y12  = (float*)alloc((size_t)Ti * 2 * D * 4);   // concat(y1,y2)
  _Float16* h_q  = (_Float16*)alloc((size_t)Tm * D * 2);
  _Float16* h_k  = (_Float16*)alloc((size_t)Tm * D * 2);
  _Float16* h_vt = (_Float16*)alloc((size_t)Tm * D * 2);

  auto LN = [&](const float* x, int T, const float* g, const float* bb, float* o) {
    ln768_kernel<<<dim3(T), dim3(256), 0, stream>>>(x, g, bb, o);
  };
  auto GEMM = [&](const float* A, int lda, const float* W, const float* bias,
                  const float* res, int ldres, float* C, int ldc, int M, int N,
                  int K, int gelu) {
    const dim3 g(N / 128, M / 128), blk(256);
    if (gelu)
      gemm_wmma_kernel<true, false, true><<<g, blk, 0, stream>>>(
          A, lda, W, bias, res, ldres, C, ldc, K);
    else if (bias && res)
      gemm_wmma_kernel<true, true, false><<<g, blk, 0, stream>>>(
          A, lda, W, bias, res, ldres, C, ldc, K);
    else if (bias)
      gemm_wmma_kernel<true, false, false><<<g, blk, 0, stream>>>(
          A, lda, W, bias, res, ldres, C, ldc, K);
    else
      gemm_wmma_kernel<false, false, false><<<g, blk, 0, stream>>>(
          A, lda, W, bias, res, ldres, C, ldc, K);
  };
  auto R2D = [&](const float* src, int ld, const int* pos, _Float16* dst, int S) {
    rope2d_pack_kernel<<<dim3(B * S), dim3(384), 0, stream>>>(src, ld, pos, dst, S);
  };
  auto R1D = [&](const float* src, int ld, _Float16* dst, int S) {
    rope1d_pack_kernel<<<dim3(B * S), dim3(384), 0, stream>>>(src, ld, dst, S);
  };
  auto PVT = [&](const float* src, int ld, _Float16* dst, int S) {
    pack_vt_kernel<<<dim3(B * S), dim3(768), 0, stream>>>(src, ld, dst, S);
  };
  auto ATTN = [&](const _Float16* q, const _Float16* k, const _Float16* vt,
                  float* o, int Sq, int Sk) {
    flash_attn_kernel<<<dim3(Sq / 128, B * N_HEADS), dim3(256), 0, stream>>>(
        q, k, vt, o, D, 0, Sq, Sk);
  };

  // norm_y on unmasked image tokens (used by ca1/ca3)
  LN(unmasked, Tu, PF(5), PF(6), f_unm);

  // ---- self_attn1: masked img, RoPE2D ----
  LN(masked, Tm, PF(7), PF(8), f_ln);
  GEMM(f_ln, D, PF(19), nullptr, nullptr, 0, f_big, 3 * D, Tm, 3 * D, D, 0);
  R2D(f_big, 3 * D, mpos, h_q, Sm);
  R2D(f_big + D, 3 * D, mpos, h_k, Sm);
  PVT(f_big + 2 * D, 3 * D, h_vt, Sm);
  ATTN(h_q, h_k, h_vt, f_ao, Sm, Sm);
  GEMM(f_ao, D, PF(20), PF(21), masked, D, out_m2, D, Tm, D, D, 0);  // masked2

  // ---- self_attn2: imu, RoPE1D ----
  LN(imu, Ti, PF(9), PF(10), f_ln);
  GEMM(f_ln, D, PF(22), nullptr, nullptr, 0, f_big, 3 * D, Ti, 3 * D, D, 0);
  R1D(f_big, 3 * D, h_q, Si);
  R1D(f_big + D, 3 * D, h_k, Si);
  PVT(f_big + 2 * D, 3 * D, h_vt, Si);
  ATTN(h_q, h_k, h_vt, f_ao, Si, Si);
  GEMM(f_ao, D, PF(23), PF(24), imu, D, f_imu2, D, Ti, D, D, 0);     // imu2

  LN(out_m2, Tm, PF(11), PF(12), f_mn);   // mn
  LN(f_imu2, Ti, PF(13), PF(14), f_inn);  // inn

  // ---- cross_attn1: masked -> unmasked (both RoPE2D) ----
  GEMM(f_mn, D, PF(25), nullptr, nullptr, 0, f_p1, D, Tm, D, D, 0);
  R2D(f_p1, D, mpos, h_q, Sm);
  GEMM(f_unm, D, PF(26), nullptr, nullptr, 0, f_p2, D, Tu, D, D, 0);
  R2D(f_p2, D, upos, h_k, Su);
  GEMM(f_unm, D, PF(27), nullptr, nullptr, 0, f_p2, D, Tu, D, D, 0);
  PVT(f_p2, D, h_vt, Su);
  ATTN(h_q, h_k, h_vt, f_ao, Sm, Su);
  GEMM(f_ao, D, PF(28), PF(29), nullptr, 0, f_x12, 2 * D, Tm, D, D, 0);       // x1

  // ---- cross_attn2: masked -> imu (k uses projq weight) ----
  GEMM(f_mn, D, PF(30), nullptr, nullptr, 0, f_p1, D, Tm, D, D, 0);
  R2D(f_p1, D, mpos, h_q, Sm);
  GEMM(f_inn, D, PF(30), nullptr, nullptr, 0, f_p2, D, Ti, D, D, 0);
  R1D(f_p2, D, h_k, Si);
  GEMM(f_inn, D, PF(31), nullptr, nullptr, 0, f_p2, D, Ti, D, D, 0);
  PVT(f_p2, D, h_vt, Si);
  ATTN(h_q, h_k, h_vt, f_ao, Sm, Si);
  GEMM(f_ao, D, PF(32), PF(33), nullptr, 0, f_x12 + D, 2 * D, Tm, D, D, 0);   // x2

  // fuse1 (+masked2 residual), then MLP1 with residual
  GEMM(f_x12, 2 * D, PF(42), PF(43), out_m2, D, out_x, D, Tm, D, 2 * D, 0);
  LN(out_x, Tm, PF(15), PF(16), f_ln);
  GEMM(f_ln, D, PF(46), PF(47), nullptr, 0, f_big, HID, Tm, HID, D, 1);       // GELU
  GEMM(f_big, HID, PF(48), PF(49), out_x, D, out_x, D, Tm, D, HID, 0);

  // ---- cross_attn3: imu -> unmasked (k uses projq weight) ----
  GEMM(f_inn, D, PF(34), nullptr, nullptr, 0, f_p2, D, Ti, D, D, 0);
  R1D(f_p2, D, h_q, Si);
  GEMM(f_unm, D, PF(34), nullptr, nullptr, 0, f_p2, D, Tu, D, D, 0);
  R2D(f_p2, D, upos, h_k, Su);
  GEMM(f_unm, D, PF(35), nullptr, nullptr, 0, f_p2, D, Tu, D, D, 0);
  PVT(f_p2, D, h_vt, Su);
  ATTN(h_q, h_k, h_vt, f_ao, Si, Su);
  GEMM(f_ao, D, PF(36), PF(37), nullptr, 0, f_y12, 2 * D, Ti, D, D, 0);       // y1

  // ---- cross_attn4: imu -> masked (k uses projq weight) ----
  GEMM(f_inn, D, PF(38), nullptr, nullptr, 0, f_p2, D, Ti, D, D, 0);
  R1D(f_p2, D, h_q, Si);
  GEMM(f_mn, D, PF(38), nullptr, nullptr, 0, f_p1, D, Tm, D, D, 0);
  R2D(f_p1, D, mpos, h_k, Sm);
  GEMM(f_mn, D, PF(39), nullptr, nullptr, 0, f_p1, D, Tm, D, D, 0);
  PVT(f_p1, D, h_vt, Sm);
  ATTN(h_q, h_k, h_vt, f_ao, Si, Sm);
  GEMM(f_ao, D, PF(40), PF(41), nullptr, 0, f_y12 + D, 2 * D, Ti, D, D, 0);   // y2

  // fuse2 (+imu2 residual), then MLP2 with residual
  GEMM(f_y12, 2 * D, PF(44), PF(45), f_imu2, D, out_y, D, Ti, D, 2 * D, 0);
  LN(out_y, Ti, PF(17), PF(18), f_ln);
  GEMM(f_ln, D, PF(50), PF(51), nullptr, 0, f_big, HID, Ti, HID, D, 1);       // GELU
  GEMM(f_big, HID, PF(52), PF(53), out_y, D, out_y, D, Ti, D, HID, 0);
#undef PF
}
